// GoSpecificWattiPooling_88923002896470
// MI455X (gfx1250) — matrix operand order repaired and checked by reading, not compiled, not run on hardware
//
#include <hip/hip_runtime.h>

// ---------------------------------------------------------------------------
// Types / helpers
// ---------------------------------------------------------------------------
typedef __attribute__((ext_vector_type(16))) __bf16          v16bf;
typedef __attribute__((ext_vector_type(8)))  float           v8f;
typedef __attribute__((ext_vector_type(8)))  unsigned short  ushortx8;

union Frag16 {
  ushortx8       h[2];
  v16bf          v;
  unsigned short u[16];
};

__device__ __forceinline__ unsigned short f2bf(float f) {
  return __builtin_bit_cast(unsigned short, (__bf16)f);   // native fptrunc, RNE
}

__device__ __forceinline__ v8f wmma_bf16(v16bf a, v16bf b, v8f c) {
  // D = A(16x32 bf16) * B(32x16 bf16) + C(16x16 f32)
  return __builtin_amdgcn_wmma_f32_16x16x32_bf16(false, a, false, b,
                                                 (short)0, c, false, false);
}

__device__ __forceinline__ float rmax16(float v) {
  v = fmaxf(v, __shfl_xor(v, 1, 32));
  v = fmaxf(v, __shfl_xor(v, 2, 32));
  v = fmaxf(v, __shfl_xor(v, 4, 32));
  v = fmaxf(v, __shfl_xor(v, 8, 32));
  return v;
}
__device__ __forceinline__ float rsum16(float v) {
  v += __shfl_xor(v, 1, 32);
  v += __shfl_xor(v, 2, 32);
  v += __shfl_xor(v, 4, 32);
  v += __shfl_xor(v, 8, 32);
  return v;
}

#define B_   4
#define L_   4096
#define T_   1024
#define DH_  1280
#define DG_  768
#define P_   256

// ---------------------------------------------------------------------------
// Kernel 1: f32 -> bf16 conversion, 4 elements / thread
// ---------------------------------------------------------------------------
__global__ void __launch_bounds__(256)
convert_bf16_kernel(const float* __restrict__ x, unsigned short* __restrict__ y, int n4) {
  int i = blockIdx.x * 256 + threadIdx.x;
  if (i >= n4) return;
  float4 f = ((const float4*)x)[i];
  unsigned long long p = (unsigned long long)f2bf(f.x)
                       | ((unsigned long long)f2bf(f.y) << 16)
                       | ((unsigned long long)f2bf(f.z) << 32)
                       | ((unsigned long long)f2bf(f.w) << 48);
  ((unsigned long long*)y)[i] = p;
}

// ---------------------------------------------------------------------------
// Kernel 2: projection GEMM on pre-converted bf16 operands.
//   Y[r][p] = bf16( sum_d X[r][d] * W[p][d] )
//   Xb: [R, D] bf16, Wb: [256, D] bf16, Y: [R, 256] bf16
//   Grid: R/16 workgroups x 256 threads (8 waves). Wave w owns p-tiles 2w,2w+1.
//   Inner loop: 6 x global_load_b128 + 2 x v_wmma per k-step, no conversions.
// ---------------------------------------------------------------------------
__global__ void __launch_bounds__(256)
proj_bf16_kernel(const unsigned short* __restrict__ Xb,
                 const unsigned short* __restrict__ Wb,
                 unsigned short* __restrict__ Y, int D) {
  const int lane = threadIdx.x & 31;
  const int wave = threadIdx.x >> 5;
  const int r0   = blockIdx.x * 16;
  const int row  = lane & 15;
  const int s0   = (lane < 16) ? 0 : 8;    // unused; A also starts at kb
  const int kb   = (lane < 16) ? 0 : 16;   // B-fragment K-run start

  const unsigned short* xrow = Xb + (size_t)(r0 + row) * D;
  const int nt0 = wave * 2, nt1 = wave * 2 + 1;
  const unsigned short* wrow0 = Wb + (size_t)(nt0 * 16 + row) * D + kb;
  const unsigned short* wrow1 = Wb + (size_t)(nt1 * 16 + row) * D + kb;

  v8f acc0, acc1;
  #pragma unroll
  for (int v = 0; v < 8; ++v) { acc0[v] = 0.f; acc1[v] = 0.f; }

  for (int k0 = 0; k0 < D; k0 += 32) {
    Frag16 a;   // A: rows of X; K-split 0-15 (lanes 0-15) vs 16-31
    a.h[0] = *(const ushortx8*)(xrow + k0 + kb);
    a.h[1] = *(const ushortx8*)(xrow + k0 + kb + 8);
    Frag16 b0, b1;  // B[k][n] = W[p0+n][k0+k]: 16 contiguous bf16 / lane
    b0.h[0] = *(const ushortx8*)(wrow0 + k0);
    b0.h[1] = *(const ushortx8*)(wrow0 + k0 + 8);
    b1.h[0] = *(const ushortx8*)(wrow1 + k0);
    b1.h[1] = *(const ushortx8*)(wrow1 + k0 + 8);
    acc0 = wmma_bf16(a.v, b0.v, acc0);
    acc1 = wmma_bf16(a.v, b1.v, acc1);
  }

  unsigned short* yb = Y + (size_t)r0 * P_;
  #pragma unroll
  for (int v = 0; v < 8; ++v) {
    int m = (lane < 16) ? v : v + 8;
    yb[(size_t)m * P_ + nt0 * 16 + row] = f2bf(acc0[v]);
    yb[(size_t)m * P_ + nt1 * 16 + row] = f2bf(acc1[v]);
  }
}

// ---------------------------------------------------------------------------
// Kernel 3: flash attention. 1 WG per 16 T-rows, 256 threads (8 waves).
//   Wave 0: S = Q K^T (bf16 WMMA, K fragments straight from global Kb),
//           mask + scale + online softmax, publish P/cf via LDS.
//   All waves: Z slice (16x160) += P * H-chunk, H staged to LDS with
//   global_load_async_to_lds_b128 (two 640-wide d phases).
// ---------------------------------------------------------------------------
__global__ void __launch_bounds__(256)
attn_kernel(const unsigned short* __restrict__ Qb,
            const unsigned short* __restrict__ Kb,
            const unsigned short* __restrict__ Hb,
            const int* __restrict__ mask,
            float* __restrict__ out) {
  __shared__ unsigned short Hs[32][640];   // 40 KB: H chunk (bf16), one d-phase
  __shared__ unsigned short Ps[16][32];    // 1 KB: P tile (bf16)
  __shared__ float cfS[16];
  __shared__ float lsumS[16];

  const int tid   = threadIdx.x;
  const int lane  = tid & 31;
  const int waveu = __builtin_amdgcn_readfirstlane(tid >> 5); // uniform -> no EXEC masking around WMMA
  const int b     = blockIdx.x >> 6;
  const int t0    = (blockIdx.x & 63) * 16;
  const int row   = lane & 15;
  const int s0    = (lane < 16) ? 0 : 8;
  const int kb16  = (lane < 16) ? 0 : 16;

  // Wave 0 keeps all Q A-fragments (16x256) resident in VGPRs.
  v16bf qa[8];
  if (waveu == 0) {
    const unsigned short* qrow = Qb + ((size_t)b * T_ + t0 + row) * P_;
    #pragma unroll
    for (int kk = 0; kk < 8; ++kk) {
      Frag16 f;
      f.h[0] = *(const ushortx8*)(qrow + kk * 32 + kb16);
      f.h[1] = *(const ushortx8*)(qrow + kk * 32 + kb16 + 8);
      qa[kk] = f.v;
    }
  }

  v8f acc[10];                       // 16 x 160 f32 Z slice (10 n-tiles)
  #pragma unroll
  for (int t = 0; t < 10; ++t)
    #pragma unroll
    for (int v = 0; v < 8; ++v) acc[t][v] = 0.f;

  float m_run[8], l_run[8];
  #pragma unroll
  for (int v = 0; v < 8; ++v) { m_run[v] = -1e30f; l_run[v] = 0.f; }

  const int srow = tid >> 3;         // 0..31 : H-chunk row this thread stages
  const int scol = tid & 7;          // 0..7  : 16B column group

  for (int chunk = 0; chunk < L_ / 32; ++chunk) {
    const int l0 = chunk * 32;
    __syncthreads();                               // prior chunk's LDS reads done

    // ---- stage H chunk, d-phase 0 (async DMA to LDS) ----
    {
      const unsigned short* src = Hb + ((size_t)b * L_ + l0 + srow) * DH_;
      unsigned short* dstrow = &Hs[srow][0];
      #pragma unroll
      for (int j = 0; j < 10; ++j) {
        int c16 = scol + j * 8;
        unsigned ldsa = (unsigned)(unsigned long long)(dstrow + c16 * 8);
        unsigned long long ga = (unsigned long long)(src + c16 * 8);
        asm volatile("global_load_async_to_lds_b128 %0, %1, off"
                     :: "v"(ldsa), "v"(ga) : "memory");
      }
    }

    // ---- wave 0: S = Q K^T, mask, online softmax ----
    if (waveu == 0) {
      v8f sA, sB;
      #pragma unroll
      for (int v = 0; v < 8; ++v) { sA[v] = 0.f; sB[v] = 0.f; }
      const unsigned short* krowA = Kb + ((size_t)b * L_ + l0 + row) * P_ + kb16;
      const unsigned short* krowB = Kb + ((size_t)b * L_ + l0 + 16 + row) * P_ + kb16;
      #pragma unroll
      for (int kk = 0; kk < 8; ++kk) {
        Frag16 fb;
        fb.h[0] = *(const ushortx8*)(krowA + kk * 32);
        fb.h[1] = *(const ushortx8*)(krowA + kk * 32 + 8);
        sA = wmma_bf16(qa[kk], fb.v, sA);
        fb.h[0] = *(const ushortx8*)(krowB + kk * 32);
        fb.h[1] = *(const ushortx8*)(krowB + kk * 32 + 8);
        sB = wmma_bf16(qa[kk], fb.v, sB);
      }

      const int mA = mask[b * L_ + l0 + row];
      const int mB = mask[b * L_ + l0 + 16 + row];
      const float scale = 0.0625f;                 // 1/sqrt(256)
      #pragma unroll
      for (int v = 0; v < 8; ++v) {
        float a = mA ? -1e30f : sA[v] * scale;
        float c = mB ? -1e30f : sB[v] * scale;
        float mx   = rmax16(fmaxf(a, c));
        float mnew = fmaxf(m_run[v], mx);
        float cf   = __expf(m_run[v] - mnew);
        float pa   = __expf(a - mnew);
        float pb   = __expf(c - mnew);
        float rs   = rsum16(pa + pb);
        l_run[v] = l_run[v] * cf + rs;
        m_run[v] = mnew;
        int mrow = (lane < 16) ? v : v + 8;
        Ps[mrow][row]      = f2bf(pa);
        Ps[mrow][16 + row] = f2bf(pb);
        if (row == v) cfS[mrow] = cf;
      }
    }

    asm volatile("s_wait_asynccnt 0x0" ::: "memory");
    __syncthreads();                               // Hs phase 0 + Ps/cfS visible

    // ---- all waves: rescale accumulators, load P A-fragment ----
    float cfv[8];
    #pragma unroll
    for (int v = 0; v < 8; ++v) cfv[v] = cfS[(lane < 16) ? v : v + 8];
    #pragma unroll
    for (int t = 0; t < 10; ++t)
      #pragma unroll
      for (int v = 0; v < 8; ++v) acc[t][v] *= cfv[v];

    Frag16 pf;
    pf.h[0] = *(const ushortx8*)(&Ps[row][kb16]);
    pf.h[1] = *(const ushortx8*)(&Ps[row][kb16 + 8]);

    // ---- Z += P * H, phase 0 (d in [0,640)) ----
    #pragma unroll
    for (int i = 0; i < 5; ++i) {
      int dl = (waveu * 5 + i) * 16;
      Frag16 hb;
      #pragma unroll
      for (int j = 0; j < 16; ++j) hb.u[j] = Hs[kb16 + j][dl + row];
      acc[i] = wmma_bf16(pf.v, hb.v, acc[i]);
    }
    __syncthreads();                               // phase-0 Hs reads done

    // ---- stage H chunk, d-phase 1 ----
    {
      const unsigned short* src = Hb + ((size_t)b * L_ + l0 + srow) * DH_ + 640;
      unsigned short* dstrow = &Hs[srow][0];
      #pragma unroll
      for (int j = 0; j < 10; ++j) {
        int c16 = scol + j * 8;
        unsigned ldsa = (unsigned)(unsigned long long)(dstrow + c16 * 8);
        unsigned long long ga = (unsigned long long)(src + c16 * 8);
        asm volatile("global_load_async_to_lds_b128 %0, %1, off"
                     :: "v"(ldsa), "v"(ga) : "memory");
      }
    }
    asm volatile("s_wait_asynccnt 0x0" ::: "memory");
    __syncthreads();                               // Hs phase 1 visible

    // ---- Z += P * H, phase 1 (d in [640,1280)) ----
    #pragma unroll
    for (int i = 0; i < 5; ++i) {
      int dl = (waveu * 5 + i) * 16;
      Frag16 hb;
      #pragma unroll
      for (int j = 0; j < 16; ++j) hb.u[j] = Hs[kb16 + j][dl + row];
      acc[5 + i] = wmma_bf16(pf.v, hb.v, acc[5 + i]);
    }
  }

  __syncthreads();
  if (waveu == 0) {
    #pragma unroll
    for (int v = 0; v < 8; ++v) {
      int mrow = (lane < 16) ? v : v + 8;
      if (row == v) lsumS[mrow] = l_run[v];
    }
  }
  __syncthreads();

  float inv[8];
  #pragma unroll
  for (int v = 0; v < 8; ++v) inv[v] = 1.0f / lsumS[(lane < 16) ? v : v + 8];
  #pragma unroll
  for (int p = 0; p < 2; ++p)
    #pragma unroll
    for (int i = 0; i < 5; ++i) {
      int d0 = p * 640 + (waveu * 5 + i) * 16 + row;
      #pragma unroll
      for (int v = 0; v < 8; ++v) {
        int mrow = (lane < 16) ? v : v + 8;
        out[((size_t)b * T_ + t0 + mrow) * DH_ + d0] = acc[p * 5 + i][v] * inv[v];
      }
    }
}

// ---------------------------------------------------------------------------
// Launcher
// ---------------------------------------------------------------------------
extern "C" void kernel_launch(void* const* d_in, const int* in_sizes, int n_in,
                              void* d_out, int out_size, void* d_ws, size_t ws_size,
                              hipStream_t stream) {
  (void)in_sizes; (void)n_in; (void)out_size; (void)ws_size;
  const float* H   = (const float*)d_in[0];
  const float* G   = (const float*)d_in[1];
  const int*   msk = (const int*)d_in[2];
  const float* Wk  = (const float*)d_in[3];
  const float* Wq  = (const float*)d_in[4];
  float* out = (float*)d_out;

  // Workspace layout (bf16 elements): Hb | Kb | Qb | Gb | Wkb | Wqb  (~57 MB)
  unsigned short* Hb  = (unsigned short*)d_ws;
  unsigned short* Kb  = Hb  + (size_t)B_ * L_ * DH_;   // 20,971,520 elems
  unsigned short* Qb  = Kb  + (size_t)B_ * L_ * P_;    //  4,194,304 elems
  unsigned short* Gb  = Qb  + (size_t)B_ * T_ * P_;    //  1,048,576 elems
  unsigned short* Wkb = Gb  + (size_t)B_ * T_ * DG_;   //  3,145,728 elems
  unsigned short* Wqb = Wkb + (size_t)P_ * DH_;        //    327,680 elems

  // 1) one-time f32 -> bf16 conversions
  {
    int n4;
    n4 = (B_ * L_ * DH_) / 4;   // H  : 5,242,880
    convert_bf16_kernel<<<dim3(n4 / 256), dim3(256), 0, stream>>>(H, Hb, n4);
    n4 = (B_ * T_ * DG_) / 4;   // G  :   786,432
    convert_bf16_kernel<<<dim3(n4 / 256), dim3(256), 0, stream>>>(G, Gb, n4);
    n4 = (P_ * DH_) / 4;        // Wk :    81,920
    convert_bf16_kernel<<<dim3(n4 / 256), dim3(256), 0, stream>>>(Wk, Wkb, n4);
    n4 = (P_ * DG_) / 4;        // Wq :    49,152
    convert_bf16_kernel<<<dim3(n4 / 256), dim3(256), 0, stream>>>(Wq, Wqb, n4);
  }

  // 2) projections (pure bf16 WMMA GEMMs)
  proj_bf16_kernel<<<dim3((B_ * T_) / 16), dim3(256), 0, stream>>>(Gb, Wqb, Qb, DG_);
  proj_bf16_kernel<<<dim3((B_ * L_) / 16), dim3(256), 0, stream>>>(Hb, Wkb, Kb, DH_);

  // 3) flash attention
  attn_kernel<<<dim3(B_ * (T_ / 16)), dim3(256), 0, stream>>>(Qb, Kb, Hb, msk, out);
}
